// TransformerModel_29703993819853
// MI455X (gfx1250) — matrix operand order repaired and checked by reading, not compiled, not run on hardware
//
#include <hip/hip_runtime.h>

typedef __attribute__((ext_vector_type(16))) _Float16 v16h;
typedef __attribute__((ext_vector_type(8)))  _Float16 v8h;
typedef __attribute__((ext_vector_type(4)))  _Float16 v4h;
typedef __attribute__((ext_vector_type(8)))  float    v8f;
typedef __attribute__((ext_vector_type(4)))  int      v4i;

// Problem constants (match reference)
#define S_  500
#define B_  32
#define IN_ 64
#define H_  512
#define NH_ 4
#define FT_ 64
#define NL_ 3
#define DH_ (H_ / NH_)       // 128
#define M_  (S_ * B_)        // 16000 tokens
#define LN_EPS 1e-5f

// GEMM tiling
#define BM 128
#define BN 128
#define BK 32
#define LDT 40   // LDS row stride in halfs = 80B (16B aligned, conflict-breaking)

#if __has_builtin(__builtin_amdgcn_global_load_async_to_lds_b128)
#define HAVE_ASYNC_LDS 1
#endif

union V16U { v16h v; v8h h[2]; };

__device__ __forceinline__ void wait_async_copies() {
#ifdef HAVE_ASYNC_LDS
#if __has_builtin(__builtin_amdgcn_s_wait_asynccnt)
    __builtin_amdgcn_s_wait_asynccnt(0);
#else
    asm volatile("s_wait_asynccnt 0x0" ::: "memory");
#endif
#endif
}

__device__ __forceinline__ void cp16_async(_Float16* l, const _Float16* g) {
#ifdef HAVE_ASYNC_LDS
    typedef __attribute__((address_space(1))) v4i* gp_t;
    typedef __attribute__((address_space(3))) v4i* lp_t;
    __builtin_amdgcn_global_load_async_to_lds_b128(
        (gp_t)(__attribute__((address_space(1))) void*)(void*)g,
        (lp_t)(__attribute__((address_space(3))) void*)(void*)l, 0, 0);
#endif
}

struct StageRegs {
#ifndef HAVE_ASYNC_LDS
    v8h a[2];
    v8h b[2];
#endif
};

// Phase 1: initiate staging of a (BM x BK) A tile + (BN x BK) B tile.
// Async path: fire global_load_async_to_lds (OOB rows get ds-stored zeros).
// Fallback: issue global loads into registers (no waits yet).
template <bool FULLN>
__device__ __forceinline__ void stage_issue(StageRegs& sr,
    _Float16* sA, _Float16* sB,
    const _Float16* __restrict__ gA, const _Float16* __restrict__ gW,
    int tid, int mBase, int nBase, int Ndim, int Kdim, int kt)
{
    const v8h z = {(_Float16)0,(_Float16)0,(_Float16)0,(_Float16)0,
                   (_Float16)0,(_Float16)0,(_Float16)0,(_Float16)0};
#ifdef HAVE_ASYNC_LDS
    #pragma unroll
    for (int i = 0; i < 2; ++i) {
        int lin = tid + i * 256, row = lin >> 2, c = lin & 3;   // 128 rows x 4 chunks
        cp16_async(sA + row * LDT + c * 8,
                   gA + (size_t)(mBase + row) * Kdim + kt + c * 8);
    }
    #pragma unroll
    for (int i = 0; i < 2; ++i) {
        int lin = tid + i * 256, row = lin >> 2, c = lin & 3;
        _Float16* l = sB + row * LDT + c * 8;
        if (FULLN) {
            cp16_async(l, gW + (size_t)(nBase + row) * Kdim + kt + c * 8);
        } else {
            int n = nBase + row;
            if (n < Ndim) cp16_async(l, gW + (size_t)n * Kdim + kt + c * 8);
            else          *(v8h*)l = z;
        }
    }
    (void)sr;
#else
    #pragma unroll
    for (int i = 0; i < 2; ++i) {
        int lin = tid + i * 256, row = lin >> 2, c = lin & 3;
        sr.a[i] = *(const v8h*)(gA + (size_t)(mBase + row) * Kdim + kt + c * 8);
    }
    #pragma unroll
    for (int i = 0; i < 2; ++i) {
        int lin = tid + i * 256, row = lin >> 2, c = lin & 3;
        int n = nBase + row;
        sr.b[i] = (FULLN || n < Ndim)
                  ? *(const v8h*)(gW + (size_t)n * Kdim + kt + c * 8) : z;
    }
#endif
}

// Phase 3: commit staged data to LDS (fallback only; async already targets LDS).
__device__ __forceinline__ void stage_commit(StageRegs& sr,
    _Float16* sA, _Float16* sB, int tid)
{
#ifndef HAVE_ASYNC_LDS
    #pragma unroll
    for (int i = 0; i < 2; ++i) {
        int lin = tid + i * 256, row = lin >> 2, c = lin & 3;
        *(v8h*)(sA + row * LDT + c * 8) = sr.a[i];
    }
    #pragma unroll
    for (int i = 0; i < 2; ++i) {
        int lin = tid + i * 256, row = lin >> 2, c = lin & 3;
        *(v8h*)(sB + row * LDT + c * 8) = sr.b[i];
    }
#else
    (void)sr; (void)sA; (void)sB; (void)tid;
#endif
}

// ---------------- Generic GEMM: C = A[M,K](f16) * W[N,K](f16)^T + bias --------
// f16 WMMA 16x16x32, fp32 accumulate; double-buffered LDS; async global->LDS
// staging; 8 waves (4Mx2N), each wave 32x64 (2x4 accumulators).
// Compile-time variants: FULLN (no N bounds checks), RELU, O32/O16 outputs.
template <bool FULLN, bool RELU, bool O32, bool O16>
__global__ __launch_bounds__(256) void gemm_wmma_f16(
    const _Float16* __restrict__ A, const _Float16* __restrict__ W,
    const float* __restrict__ bias, float* __restrict__ C32,
    _Float16* __restrict__ C16, int Ndim, int Kdim)
{
    __shared__ __align__(16) _Float16 As[2 * BM * LDT];
    __shared__ __align__(16) _Float16 Bs[2 * BN * LDT];

    const int tid  = threadIdx.x;
    const int lane = tid & 31;
    const int wave = tid >> 5;
    const int wm   = wave & 3;   // 0..3 -> M sub-block of 32
    const int wn   = wave >> 2;  // 0..1 -> N sub-block of 64
    const int mBase = blockIdx.x * BM;
    const int nBase = blockIdx.y * BN;

    const int l16     = lane & 15;
    const int halfsel = lane >> 4;
    const int aK0     = halfsel * 8;      // A-frag K origin per ISA layout
    const int bK0     = halfsel * 8;      // B-frag K origin per ISA layout

    v8f acc[2][4];
    const v8f vz = {0.f,0.f,0.f,0.f,0.f,0.f,0.f,0.f};
    for (int i = 0; i < 2; ++i)
        for (int j = 0; j < 4; ++j) acc[i][j] = vz;

    const int ntile = Kdim / BK;

    // prologue: stage tile 0 into buffer 0
    {
        StageRegs sr;
        stage_issue<FULLN>(sr, As, Bs, A, W, tid, mBase, nBase, Ndim, Kdim, 0);
        stage_commit(sr, As, Bs, tid);
        wait_async_copies();
        __syncthreads();
    }

    for (int t = 0; t < ntile; ++t) {
        const int cur = t & 1;
        _Float16* sAc = As + cur * (BM * LDT);
        _Float16* sBc = Bs + cur * (BN * LDT);
        _Float16* sAn = As + (1 - cur) * (BM * LDT);
        _Float16* sBn = Bs + (1 - cur) * (BN * LDT);
        const bool have_next = (t + 1) < ntile;

        StageRegs sr;
        if (have_next)
            stage_issue<FULLN>(sr, sAn, sBn, A, W, tid, mBase, nBase, Ndim, Kdim,
                               (t + 1) * BK);

        // ---- fragments per CDNA5 VGPR layouts ----
        V16U af[2], bf[4];
        #pragma unroll
        for (int mi = 0; mi < 2; ++mi) {
            const _Float16* p = &sAc[(wm * 32 + mi * 16 + l16) * LDT + aK0];
            af[mi].h[0] = *(const v8h*)p;          // K = k0 .. k0+7
            af[mi].h[1] = *(const v8h*)(p + 16);   // K = k0+16 .. k0+23
        }
        #pragma unroll
        for (int ni = 0; ni < 4; ++ni) {
            const _Float16* p = &sBc[(wn * 64 + ni * 16 + l16) * LDT + bK0];
            bf[ni].h[0] = *(const v8h*)p;          // K = k0 .. k0+7
            bf[ni].h[1] = *(const v8h*)(p + 16);   // K = k0+16 .. k0+23
        }
        #pragma unroll
        for (int mi = 0; mi < 2; ++mi)
            #pragma unroll
            for (int ni = 0; ni < 4; ++ni)
                acc[mi][ni] = __builtin_amdgcn_wmma_f32_16x16x32_f16(
                    false, af[mi].v, false, bf[ni].v,
                    (short)0, acc[mi][ni], false, false);

        if (have_next) {
            stage_commit(sr, sAn, sBn, tid);
            wait_async_copies();
            __syncthreads();
        }
    }

    // ---- epilogue: bias (+ReLU), store per C/D layout ----
    #pragma unroll
    for (int mi = 0; mi < 2; ++mi) {
        #pragma unroll
        for (int ni = 0; ni < 4; ++ni) {
            int col = nBase + wn * 64 + ni * 16 + l16;
            if (FULLN || col < Ndim) {
                float bv = bias[col];
                #pragma unroll
                for (int r = 0; r < 8; ++r) {
                    int mrow = mBase + wm * 32 + mi * 16 + (halfsel ? r + 8 : r);
                    float v = acc[mi][ni][r] + bv;
                    if (RELU) v = fmaxf(v, 0.f);
                    if (O32) C32[(size_t)mrow * Ndim + col] = v;
                    if (O16) C16[(size_t)mrow * Ndim + col] = (_Float16)v;
                }
            }
        }
    }
}

// ---------------- fp32 -> f16 conversion (vectorized x4) ----------------------
__global__ __launch_bounds__(256) void cvt_f32_f16(
    const float* __restrict__ s, _Float16* __restrict__ d, int n4)
{
    int i = blockIdx.x * 256 + threadIdx.x;
    if (i < n4) {
        float4 v = ((const float4*)s)[i];
        v4h h = { (_Float16)v.x, (_Float16)v.y, (_Float16)v.z, (_Float16)v.w };
        ((v4h*)d)[i] = h;
    }
}

// ---------------- Band-2 attention: attn over j in {i-1, i} -------------------
// One wave per (s, b, h). Lane owns 4 consecutive d of DH=128. Emits f16.
__global__ __launch_bounds__(256) void attn_band2(
    const float* __restrict__ qkv, _Float16* __restrict__ o)
{
    const int wid  = blockIdx.x * 8 + (threadIdx.x >> 5);
    const int lane = threadIdx.x & 31;
    if (wid >= S_ * B_ * NH_) return;
    const int h  = wid % NH_;
    const int tb = wid / NH_;
    const int s  = tb / B_;
    const size_t tok    = (size_t)tb;
    const size_t stride = 3 * H_;

    const float* q = qkv + tok * stride + h * DH_;
    const float* k = qkv + tok * stride + H_ + h * DH_;
    const float* v = qkv + tok * stride + 2 * H_ + h * DH_;

    const int d0 = lane * 4;
    float4 qv = *(const float4*)(q + d0);
    float4 kv = *(const float4*)(k + d0);
    float s1 = qv.x*kv.x + qv.y*kv.y + qv.z*kv.z + qv.w*kv.w;
    float s0 = 0.f;
    float4 vp = {0.f, 0.f, 0.f, 0.f};
    if (s > 0) {
        const float* km = k - (size_t)B_ * stride;  // token (s-1, b)
        const float* vm = v - (size_t)B_ * stride;
        float4 kp = *(const float4*)(km + d0);
        vp        = *(const float4*)(vm + d0);
        s0 = qv.x*kp.x + qv.y*kp.y + qv.z*kp.z + qv.w*kp.w;
    }
    #pragma unroll
    for (int off = 16; off; off >>= 1) {
        s1 += __shfl_xor(s1, off, 32);
        s0 += __shfl_xor(s0, off, 32);
    }
    const float scale = 0.08838834764831845f;   // 1/sqrt(128)
    s1 *= scale; s0 *= scale;
    float w0, w1;
    if (s > 0) {
        float mx  = fmaxf(s0, s1);
        float e0  = __expf(s0 - mx);
        float e1  = __expf(s1 - mx);
        float inv = 1.f / (e0 + e1);
        w0 = e0 * inv; w1 = e1 * inv;
    } else { w0 = 0.f; w1 = 1.f; }

    float4 vv = *(const float4*)(v + d0);
    v4h oh = { (_Float16)(w0*vp.x + w1*vv.x), (_Float16)(w0*vp.y + w1*vv.y),
               (_Float16)(w0*vp.z + w1*vv.z), (_Float16)(w0*vp.w + w1*vv.w) };
    *(v4h*)(o + tok * H_ + h * DH_ + d0) = oh;
}

// ---------------- Fused residual + LayerNorm: x = LN(x + r)*g + b -------------
// Emits fp32 (residual chain) and f16 (next GEMM A-operand).
__global__ __launch_bounds__(256) void resid_ln(
    float* __restrict__ x, const float* __restrict__ r,
    const float* __restrict__ gamma, const float* __restrict__ beta,
    _Float16* __restrict__ xh)
{
    const int tok  = blockIdx.x * 8 + (threadIdx.x >> 5);
    const int lane = threadIdx.x & 31;
    if (tok >= M_) return;
    float* xp = x + (size_t)tok * H_;
    const float* rp = r + (size_t)tok * H_;

    float4 xs[4];
    float sum = 0.f;
    #pragma unroll
    for (int j = 0; j < 4; ++j) {
        int idx = (lane + j * 32) * 4;
        float4 a = *(const float4*)(xp + idx);
        float4 c = *(const float4*)(rp + idx);
        a.x += c.x; a.y += c.y; a.z += c.z; a.w += c.w;
        xs[j] = a;
        sum += a.x + a.y + a.z + a.w;
    }
    #pragma unroll
    for (int off = 16; off; off >>= 1) sum += __shfl_xor(sum, off, 32);
    const float mu = sum * (1.f / H_);

    float vs = 0.f;
    #pragma unroll
    for (int j = 0; j < 4; ++j) {
        float dx = xs[j].x - mu, dy = xs[j].y - mu, dz = xs[j].z - mu, dw = xs[j].w - mu;
        vs += dx*dx + dy*dy + dz*dz + dw*dw;
    }
    #pragma unroll
    for (int off = 16; off; off >>= 1) vs += __shfl_xor(vs, off, 32);
    const float rstd = rsqrtf(vs * (1.f / H_) + LN_EPS);

    #pragma unroll
    for (int j = 0; j < 4; ++j) {
        int idx = (lane + j * 32) * 4;
        float4 g  = *(const float4*)(gamma + idx);
        float4 bb = *(const float4*)(beta + idx);
        float4 o;
        o.x = (xs[j].x - mu) * rstd * g.x + bb.x;
        o.y = (xs[j].y - mu) * rstd * g.y + bb.y;
        o.z = (xs[j].z - mu) * rstd * g.z + bb.z;
        o.w = (xs[j].w - mu) * rstd * g.w + bb.w;
        *(float4*)(xp + idx) = o;
        v4h oh = { (_Float16)o.x, (_Float16)o.y, (_Float16)o.z, (_Float16)o.w };
        *(v4h*)(xh + (size_t)tok * H_ + idx) = oh;
    }
}

// ---------------- Host orchestration ------------------------------------------
extern "C" void kernel_launch(void* const* d_in, const int* in_sizes, int n_in,
                              void* d_out, int out_size, void* d_ws, size_t ws_size,
                              hipStream_t stream) {
    const float* src   = (const float*)d_in[0];
    const float* enc_W = (const float*)d_in[1];
    const float* enc_b = (const float*)d_in[2];
    const float* qkv_W = (const float*)d_in[3];
    const float* qkv_b = (const float*)d_in[4];
    const float* out_W = (const float*)d_in[5];
    const float* out_b = (const float*)d_in[6];
    const float* ln1_s = (const float*)d_in[7];
    const float* ln1_b = (const float*)d_in[8];
    const float* ff1_W = (const float*)d_in[9];
    const float* ff1_b = (const float*)d_in[10];
    const float* ff2_W = (const float*)d_in[11];
    const float* ff2_b = (const float*)d_in[12];
    const float* ln2_s = (const float*)d_in[13];
    const float* ln2_b = (const float*)d_in[14];
    const float* fc_W  = (const float*)d_in[15];
    const float* fc_b  = (const float*)d_in[16];

    // Workspace: fp32 x | qkv | t2, then f16 x_h | t1_h | src_h | weights_h
    float* xf   = (float*)d_ws;                         // M*H
    float* qkvf = xf   + (size_t)M_ * H_;               // M*3H
    float* t2f  = qkvf + (size_t)M_ * 3 * H_;           // M*H
    _Float16* x_h    = (_Float16*)(t2f + (size_t)M_ * H_);
    _Float16* t1_h   = x_h    + (size_t)M_ * H_;
    _Float16* src_h  = t1_h   + (size_t)M_ * H_;
    _Float16* encW_h = src_h  + (size_t)M_ * IN_;
    _Float16* qkvW_h = encW_h + (size_t)H_ * IN_;
    _Float16* outW_h = qkvW_h + (size_t)NL_ * 3 * H_ * H_;
    _Float16* ff1W_h = outW_h + (size_t)NL_ * H_ * H_;
    _Float16* ff2W_h = ff1W_h + (size_t)NL_ * H_ * H_;
    _Float16* fcW_h  = ff2W_h + (size_t)NL_ * H_ * H_;

    const dim3 blk(256);
    auto cvt = [&](const float* s, _Float16* d, size_t n) {
        int n4 = (int)(n / 4);
        cvt_f32_f16<<<dim3((n4 + 255) / 256), blk, 0, stream>>>(s, d, n4);
    };
    // One-time f16 conversions (weights + src)
    cvt(src,   src_h,  (size_t)M_ * IN_);
    cvt(enc_W, encW_h, (size_t)H_ * IN_);
    cvt(qkv_W, qkvW_h, (size_t)NL_ * 3 * H_ * H_);
    cvt(out_W, outW_h, (size_t)NL_ * H_ * H_);
    cvt(ff1_W, ff1W_h, (size_t)NL_ * H_ * H_);
    cvt(ff2_W, ff2W_h, (size_t)NL_ * H_ * H_);
    cvt(fc_W,  fcW_h,  (size_t)FT_ * H_);

    // GEMM variants (compile-time specialized)
    auto gemm_f32f16 = [&](const _Float16* A, const _Float16* W, const float* bias,
                           float* C32, _Float16* C16, int N, int K) {
        dim3 grid(M_ / BM, N / BN);
        gemm_wmma_f16<true, false, true, true><<<grid, blk, 0, stream>>>(
            A, W, bias, C32, C16, N, K);
    };
    auto gemm_f32 = [&](const _Float16* A, const _Float16* W, const float* bias,
                        float* C32, int N, int K) {
        dim3 grid(M_ / BM, N / BN);
        gemm_wmma_f16<true, false, true, false><<<grid, blk, 0, stream>>>(
            A, W, bias, C32, nullptr, N, K);
    };
    auto gemm_relu_f16 = [&](const _Float16* A, const _Float16* W, const float* bias,
                             _Float16* C16, int N, int K) {
        dim3 grid(M_ / BM, N / BN);
        gemm_wmma_f16<true, true, false, true><<<grid, blk, 0, stream>>>(
            A, W, bias, nullptr, C16, N, K);
    };
    auto gemm_part_f32 = [&](const _Float16* A, const _Float16* W, const float* bias,
                             float* C32, int N, int K) {
        dim3 grid(M_ / BM, (N + BN - 1) / BN);
        gemm_wmma_f16<false, false, true, false><<<grid, blk, 0, stream>>>(
            A, W, bias, C32, nullptr, N, K);
    };

    // encode: x = src @ enc_W^T + enc_b   (fp32 + f16 outputs)
    gemm_f32f16(src_h, encW_h, enc_b, xf, x_h, H_, IN_);

    for (int l = 0; l < NL_; ++l) {
        gemm_f32(x_h, qkvW_h + (size_t)l * 3 * H_ * H_, qkv_b + (size_t)l * 3 * H_,
                 qkvf, 3 * H_, H_);
        attn_band2<<<dim3((S_ * B_ * NH_) / 8), blk, 0, stream>>>(qkvf, t1_h);
        gemm_f32(t1_h, outW_h + (size_t)l * H_ * H_, out_b + (size_t)l * H_,
                 t2f, H_, H_);
        resid_ln<<<dim3(M_ / 8), blk, 0, stream>>>(
            xf, t2f, ln1_s + (size_t)l * H_, ln1_b + (size_t)l * H_, x_h);
        gemm_relu_f16(x_h, ff1W_h + (size_t)l * H_ * H_, ff1_b + (size_t)l * H_,
                      t1_h, H_, H_);
        gemm_f32(t1_h, ff2W_h + (size_t)l * H_ * H_, ff2_b + (size_t)l * H_,
                 t2f, H_, H_);
        resid_ln<<<dim3(M_ / 8), blk, 0, stream>>>(
            xf, t2f, ln2_s + (size_t)l * H_, ln2_b + (size_t)l * H_, x_h);
    }

    gemm_part_f32(x_h, fcW_h, fc_b, (float*)d_out, FT_, H_);
}
